// SSD_33552284517034
// MI455X (gfx1250) — matrix operand order repaired and checked by reading, not compiled, not run on hardware
//
#include <hip/hip_runtime.h>
#include <stdint.h>
#include <stddef.h>

// ---------------------------------------------------------------------------
// Types for gfx1250 WMMA (wave32, 16x16x32 bf16 -> f32)
// ---------------------------------------------------------------------------
typedef __bf16 bf16_t;
typedef __attribute__((ext_vector_type(16))) bf16_t v16bf;
typedef __attribute__((ext_vector_type(8)))  bf16_t v8bf;
typedef __attribute__((ext_vector_type(8)))  float  v8f;

#if __has_builtin(__builtin_amdgcn_global_load_async_to_lds_b32) && \
    __has_builtin(__builtin_amdgcn_s_wait_asynccnt)
#define USE_ASYNC_LDS 1
typedef __attribute__((address_space(1))) int as1_int;
typedef __attribute__((address_space(3))) int as3_int;
#else
#define USE_ASYNC_LDS 0
#endif

__device__ __forceinline__ bf16_t f2bf(float x) {
  union { float f; uint32_t u; } v; v.f = x;
  uint32_t r = v.u + 0x7FFFu + ((v.u >> 16) & 1u);   // round-to-nearest-even
  uint16_t h = (uint16_t)(r >> 16);
  return __builtin_bit_cast(bf16_t, h);
}

// ---------------------------------------------------------------------------
// NT GEMM: C[m,n] = act( sum_k A[m,k]*Bt[n,k] + bias + residual )
// A : [>=pad32(M), lda] bf16 row-major (K contiguous, lda multiple of 32)
// Bt: [>=pad64(N), ldb] bf16 row-major (K contiguous, ldb multiple of 32)
// K is a multiple of 32 (operands zero-padded by the staging kernels).
//
// Each wave computes a 32x64 tile = 2(M) x 4(N) WMMA subtiles ->
// 8 back-to-back v_wmma_f32_16x16x32_bf16 per K-step fed by 12 fragment
// loads (A fragments reused 4x, B fragments 2x).  The main loop is fully
// branch-free and wave-uniform (edge waves CLAMP to the last in-bounds tile
// and recompute it), so EXEC stays all-1s around every WMMA as the ISA
// requires, with a single loadcnt wait per K-step.
//
// Fragment layout used here (wave32; A and B share contiguous k order):
//  A 16x32: lanes 0-15 row M=lane, K 0..15; lanes 16-31 row M=lane-16,
//           K 16..31 -> two 16B loads per lane.
//  B 32x16: lanes 0-15 col N=lane, K 0..15; lanes 16-31 col N=lane-16,
//           K 16..31 -> one 32B load per lane.
//  C 16x16: VGPR r -> row (+8 for upper half-wave), col = lane&15.
// ---------------------------------------------------------------------------
__global__ void k_gemm_nt(const bf16_t* __restrict__ A, const bf16_t* __restrict__ B,
                          float* __restrict__ C,
                          int M, int N, int K, int Malloc, int Nalloc,
                          int lda, int ldb, int ldc,
                          const float* __restrict__ bias, int biasMode,
                          const float* __restrict__ res, int ldres, int act,
                          long sA, long sB, long sC, long sRes)
{
  const int lane = threadIdx.x & 31;
  const int wave = threadIdx.x >> 5;        // 8 waves: 4 in M x 2 in N
  const int wy = wave >> 1, wx = wave & 1;
  const int half = lane >> 4;
  const int lr   = lane & 15;
  int m0 = (blockIdx.y * 4 + wy) * 32;
  if (m0 + 32 > Malloc) m0 = Malloc - 32;   // clamp: recompute last tile
  int n0 = (blockIdx.x * 2 + wx) * 64;
  if (n0 + 64 > Nalloc) n0 = Nalloc - 64;   // clamp: recompute last tile

  const long bz = blockIdx.z;
  const bf16_t* Ab = A + bz * sA;
  const bf16_t* Bb = B + bz * sB;
  float* Cb = C + bz * sC;
  const float* resp = res ? (res + bz * sRes) : (const float*)0;

  const bf16_t* arow0 = Ab + (long)(m0 + lr) * lda;
  const bf16_t* arow1 = arow0 + (long)16 * lda;
  const bf16_t* brow[4];
#pragma unroll
  for (int t = 0; t < 4; ++t)
    brow[t] = Bb + (long)(n0 + t * 16 + lr) * ldb;

  const v8f vzero = {};
  v8f acc[2][4];
#pragma unroll
  for (int i = 0; i < 2; ++i)
#pragma unroll
    for (int t = 0; t < 4; ++t) acc[i][t] = vzero;

  for (int k0 = 0; k0 < K; k0 += 32) {
    __builtin_prefetch(arow0 + k0 + 32, 0, 1);   // speculative, OOB-safe
    __builtin_prefetch(arow1 + k0 + 32, 0, 1);
    v16bf av[2], bv[4];
#pragma unroll
    for (int i = 0; i < 2; ++i) {
      const bf16_t* ar = (i == 0) ? arow0 : arow1;
      v8bf lo = *(const v8bf*)(ar + k0 + half * 16);       // 16B aligned
      v8bf hi = *(const v8bf*)(ar + k0 + half * 16 + 8);   // 16B aligned
#pragma unroll
      for (int e = 0; e < 8; ++e) { av[i][e] = lo[e]; av[i][e + 8] = hi[e]; }
    }
#pragma unroll
    for (int t = 0; t < 4; ++t)
      bv[t] = *(const v16bf*)(brow[t] + k0 + half * 16);   // 32B aligned
#pragma unroll
    for (int i = 0; i < 2; ++i)
#pragma unroll
      for (int t = 0; t < 4; ++t)
        acc[i][t] = __builtin_amdgcn_wmma_f32_16x16x32_bf16(
            false, av[i], false, bv[t], (short)0, acc[i][t], false, false);
  }

#pragma unroll
  for (int i = 0; i < 2; ++i) {
#pragma unroll
    for (int t = 0; t < 4; ++t) {
      int col = n0 + t * 16 + lr;
      if (col >= N) continue;
#pragma unroll
      for (int r8 = 0; r8 < 8; ++r8) {
        int row = m0 + i * 16 + r8 + half * 8;
        if (row < M) {
          float v = acc[i][t][r8];
          if (bias) v += (biasMode == 2) ? bias[col] : bias[row];
          if (resp) v += resp[(long)row * ldres + col];
          if (act)  v = v > 0.f ? v : 0.f;
          Cb[(long)row * ldc + col] = v;
        }
      }
    }
  }
}

// ---------------------------------------------------------------------------
// Staging: f32 -> bf16 with zero padding, same orientation.
// ---------------------------------------------------------------------------
__global__ void k_cast_pad(const float* __restrict__ src, bf16_t* __restrict__ dst,
                           int R, int C, int lds, int Rp, int Cp, long sSrc, long sDst)
{
  long idx = (long)blockIdx.x * blockDim.x + threadIdx.x;
  long total = (long)Rp * Cp;
  if (idx >= total) return;
  int r = (int)(idx / Cp), c = (int)(idx % Cp);
  const float* s = src + (long)blockIdx.y * sSrc;
  bf16_t* d = dst + (long)blockIdx.y * sDst;
  float v = (r < R && c < C) ? s[(long)r * lds + c] : 0.f;
  d[idx] = f2bf(v);
}

// ---------------------------------------------------------------------------
// Staging: f32 -> bf16 transposed with zero padding, LDS 16x16 tiles.
// Uses the gfx1250 async global->LDS engine when the builtin is available.
// src: [R, C] ld=lds ; dst: [dstR, dstC] with dst[c,r] = src[r,c]
// ---------------------------------------------------------------------------
__global__ void k_cast_T_pad(const float* __restrict__ src, bf16_t* __restrict__ dst,
                             int R, int C, int lds, int dstR, int dstC,
                             long sSrc, long sDst)
{
  __shared__ float tile[16][17];
  const float* s = src + (long)blockIdx.z * sSrc;
  bf16_t* d = dst + (long)blockIdx.z * sDst;
  int r0 = blockIdx.x * 16;               // src row block
  int c0 = blockIdx.y * 16;               // src col block
  int tx = threadIdx.x & 15, ty = threadIdx.x >> 4;
  int r = r0 + ty, c = c0 + tx;
  bool valid = (r < R && c < C);
#if USE_ASYNC_LDS
  tile[ty][tx] = 0.f;
  __syncthreads();
  if (valid) {
    // AS1 pointer: flat global address == AS1 address (cast via integer).
    // AS3 pointer: low 32 bits of the flat LDS address are the LDS offset.
    __builtin_amdgcn_global_load_async_to_lds_b32(
        (as1_int*)(uintptr_t)(s + (long)r * lds + c),
        (as3_int*)(uint32_t)(uintptr_t)(&tile[ty][tx]),
        0, 0);
  }
  __builtin_amdgcn_s_wait_asynccnt(0);
  __syncthreads();
#else
  tile[ty][tx] = valid ? s[(long)r * lds + c] : 0.f;
  __syncthreads();
#endif
  int dr = c0 + ty, dc = r0 + tx;
  if (dr < dstR && dc < dstC)
    d[(long)dr * dstC + dc] = f2bf(tile[tx][ty]);
}

// ---------------------------------------------------------------------------
// L2Norm over channels of q0 (SSD L2Norm(512, 20))
// ---------------------------------------------------------------------------
__global__ void k_l2norm(const float* __restrict__ q, const float* __restrict__ w,
                         float* __restrict__ out, int C, int Np, int tot)
{
  int idx = blockIdx.x * blockDim.x + threadIdx.x;   // over B*Np
  if (idx >= tot) return;
  int b = idx / Np, n = idx % Np;
  const float* src = q + (long)b * C * Np + n;
  float ss = 0.f;
  for (int c = 0; c < C; ++c) { float v = src[(long)c * Np]; ss += v * v; }
  float inv = 1.f / (sqrtf(ss) + 1e-10f);
  float* dst = out + (long)b * C * Np + n;
  for (int c = 0; c < C; ++c) dst[(long)c * Np] = src[(long)c * Np] * inv * w[c];
}

// ---------------------------------------------------------------------------
// Row softmax of scores f32 -> probabilities bf16 (A-fragment layout: row
// major, keys contiguous, zero-padded to NPAD). grid.x = NPAD rows, grid.y = b*s
// ---------------------------------------------------------------------------
__global__ void k_softmax_bf16(const float* __restrict__ f, bf16_t* __restrict__ p,
                               int Nvalid, int NPAD, long sF, long sP)
{
  __shared__ float red[256];
  int row = blockIdx.x;
  const float* fr = f + (long)blockIdx.y * sF + (long)row * NPAD;
  bf16_t* pr = p + (long)blockIdx.y * sP + (long)row * NPAD;
  int tid = threadIdx.x;
  if (row >= Nvalid) {                    // zero pad rows (block-uniform branch)
    for (int m = tid; m < NPAD; m += blockDim.x) pr[m] = f2bf(0.f);
    return;
  }
  float mx = -1e30f;
  for (int m = tid; m < Nvalid; m += blockDim.x) mx = fmaxf(mx, fr[m]);
  red[tid] = mx; __syncthreads();
  for (int s = 128; s > 0; s >>= 1) { if (tid < s) red[tid] = fmaxf(red[tid], red[tid + s]); __syncthreads(); }
  mx = red[0]; __syncthreads();
  float sum = 0.f;
  for (int m = tid; m < Nvalid; m += blockDim.x) sum += __expf(fr[m] - mx);
  red[tid] = sum; __syncthreads();
  for (int s = 128; s > 0; s >>= 1) { if (tid < s) red[tid] += red[tid + s]; __syncthreads(); }
  float inv = 1.f / red[0];
  for (int m = tid; m < NPAD; m += blockDim.x)
    pr[m] = f2bf(m < Nvalid ? __expf(fr[m] - mx) * inv : 0.f);
}

// ---------------------------------------------------------------------------
// Support GAP over spatial dims: (B*S, C, HW) -> (B*S, C)
// ---------------------------------------------------------------------------
__global__ void k_gap(const float* __restrict__ x, float* __restrict__ out,
                      int C, int HW, int tot)
{
  int i = blockIdx.x * blockDim.x + threadIdx.x;   // over B*S*C
  if (i >= tot) return;
  const float* p = x + (long)i * HW;
  float acc = 0.f;
  for (int k = 0; k < HW; ++k) acc += p[k];
  out[i] = acc / (float)HW;
}

// ---------------------------------------------------------------------------
// Per-batch GNN gating (S=5). One block per batch element; LDS ping-pong for
// the wcompute MLP activations, global scratch for x / aggregated x.
// ---------------------------------------------------------------------------
#define GNN_HMAX 192
struct GnnP {
  const float* wc_w[6][5];
  const float* wc_b[6][5];
  const float* gw[6];
  const float* gb[6];
};

__global__ void k_gnn(const float* __restrict__ gap, float* __restrict__ gates,
                      float* __restrict__ xbuf, float* __restrict__ xabuf,
                      int C, int FStride, GnnP P)
{
  __shared__ float hA[25 * GNN_HMAX];
  __shared__ float hB[25 * GNN_HMAX];
  __shared__ float Amat[25];
  __shared__ float Araw[25];
  const int b = blockIdx.x;
  const int tid = threadIdx.x;
  const int NT = blockDim.x;
  float* x  = xbuf  + (long)b * 5 * FStride;
  float* xa = xabuf + (long)b * 5 * FStride;

  for (int i = tid; i < 5 * C; i += NT) {
    int s = i / C, f = i % C;
    x[s * FStride + f] = gap[((long)b * 5 + s) * C + f];
  }
  __syncthreads();

  int F = C;
  for (int L = 0; L <= 5; ++L) {
    int dims[6] = { F, 192, 192, 96, 96, 1 };
    float* cur = hA; float* nxt = hB;
    // wcompute MLP on |x_i - x_j|
    for (int st = 0; st < 5; ++st) {
      int din = dims[st], dout = dims[st + 1];
      const float* W  = P.wc_w[L][st];
      const float* Bv = P.wc_b[L][st];
      for (int item = tid; item < 25 * dout; item += NT) {
        int pp = item / dout, o = item % dout;
        float acc = Bv[o];
        if (st == 0) {
          int i = pp / 5, j = pp % 5;
          const float* xi = x + i * FStride;
          const float* xj = x + j * FStride;
          const float* Wr = W + (long)o * din;
          for (int f = 0; f < din; ++f) acc += Wr[f] * fabsf(xi[f] - xj[f]);
        } else {
          const float* hin = cur + pp * GNN_HMAX;
          const float* Wr = W + (long)o * din;
          for (int f = 0; f < din; ++f) acc += Wr[f] * hin[f];
        }
        if (st < 4) acc = acc >= 0.f ? acc : 0.01f * acc;    // leaky
        if (st == 4) Araw[pp] = acc;
        else nxt[pp * GNN_HMAX + o] = acc;
      }
      __syncthreads();
      float* t = cur; cur = nxt; nxt = t;
    }
    // masked softmax over neighbors j
    if (tid < 5) {
      int i = tid;
      float v[5], mx = -1e30f;
      for (int j = 0; j < 5; ++j) {
        v[j] = Araw[i * 5 + j] - (i == j ? 1e8f : 0.f);
        mx = fmaxf(mx, v[j]);
      }
      float sum = 0.f;
      for (int j = 0; j < 5; ++j) { v[j] = __expf(v[j] - mx); sum += v[j]; }
      for (int j = 0; j < 5; ++j) Amat[i * 5 + j] = v[j] / sum;
    }
    __syncthreads();
    // xa = A x
    for (int item = tid; item < 5 * F; item += NT) {
      int s = item / F, f = item % F;
      float acc = 0.f;
      for (int j = 0; j < 5; ++j) acc += Amat[s * 5 + j] * x[j * FStride + f];
      xa[s * FStride + f] = acc;
    }
    __syncthreads();
    // gconv over [x ; A x]
    int gout = (L < 5) ? 48 : 1;
    const float* GW = P.gw[L]; const float* GB = P.gb[L];
    for (int item = tid; item < 5 * gout; item += NT) {
      int s = item / gout, o = item % gout;
      const float* w0 = GW + (long)o * 2 * F;
      const float* w1 = w0 + F;
      const float* xs = x + s * FStride; const float* xas = xa + s * FStride;
      float acc = GB[o];
      for (int f = 0; f < F; ++f) acc += w0[f] * xs[f] + w1[f] * xas[f];
      if (L < 5) {
        acc = acc >= 0.f ? acc : 0.01f * acc;
        x[s * FStride + F + o] = acc;
      } else {
        gates[b * 5 + s] = 1.f / (1.f + __expf(-acc));
      }
    }
    __syncthreads();
    if (L < 5) F += 48;
  }
}

// ---------------------------------------------------------------------------
// fused[c,n] = sum_s z[b,s,c,n] * gate[b,s]
// ---------------------------------------------------------------------------
__global__ void k_fused_sum(const float* __restrict__ z, const float* __restrict__ gates,
                            float* __restrict__ fused, int C, int Npos, int NP)
{
  int b = blockIdx.y;
  long idx = (long)blockIdx.x * blockDim.x + threadIdx.x;
  long tot = (long)C * Npos;
  if (idx >= tot) return;
  int c = (int)(idx / Npos), n = (int)(idx % Npos);
  float acc = 0.f;
  for (int s = 0; s < 5; ++s)
    acc += z[(((long)(b * 5 + s) * C + c) * NP) + n] * gates[b * 5 + s];
  fused[((long)b * C + c) * NP + n] = acc;
}

// ---------------------------------------------------------------------------
// conv3x3 SSD head, writing directly to the (b, box, pb) output layout.
// x: (B, Cin, H, W) with channel stride ldin.  w: (O, Cin, 3, 3)
// ---------------------------------------------------------------------------
__global__ void k_head(const float* __restrict__ x, long sx,
                       const float* __restrict__ w, const float* __restrict__ bias,
                       float* __restrict__ out,
                       int Cin, int ldin, int H, int W, int mbx, int pb,
                       int box_off, int totalBoxes)
{
  int b = blockIdx.y;
  int O = mbx * pb;
  int tot = O * H * W;
  const float* xb = x + (long)b * sx;
  for (int idx = blockIdx.x * blockDim.x + threadIdx.x; idx < tot;
       idx += gridDim.x * blockDim.x) {
    int o = idx / (H * W); int hw = idx % (H * W);
    int h = hw / W, wq = hw % W;
    float acc = bias[o];
    for (int cc = 0; cc < Cin; ++cc) {
      const float* xr = xb + (long)cc * ldin;
      const float* wr = w + ((long)o * Cin + cc) * 9;
#pragma unroll
      for (int kh = 0; kh < 3; ++kh) {
        int hh = h + kh - 1; if (hh < 0 || hh >= H) continue;
#pragma unroll
        for (int kw = 0; kw < 3; ++kw) {
          int ww = wq + kw - 1; if (ww < 0 || ww >= W) continue;
          acc += wr[kh * 3 + kw] * xr[hh * W + ww];
        }
      }
    }
    int m = o / pb, j = o % pb;
    long dst = (long)b * totalBoxes * pb + (long)(box_off + hw * mbx + m) * pb + j;
    out[dst] = acc;
  }
}

// ===========================================================================
// Host orchestration
// ===========================================================================
struct LevelParams {
  const float *conf_b, *conf_w, *fuse_b, *fuse_w;
  const float *last_gb, *last_gw;
  const float *last_wc_w[5], *last_wc_b[5];
  const float *lay_gb[5], *lay_gw[5];
  const float *lay_wc_w[5][5], *lay_wc_b[5][5];
  const float *loc_b, *loc_w;
  const float *g_b, *g_w, *phi_b, *phi_w, *theta_b, *theta_w, *wz_b, *wz_w;
};

static inline int pad32(int x) { return (x + 31) & ~31; }
static inline int pad64(int x) { return (x + 63) & ~63; }

static void gemm(hipStream_t st, const bf16_t* A, const bf16_t* B, float* C,
                 int M, int N, int K, int lda, int ldb, int ldc,
                 const float* bias, int biasMode, const float* res, int ldres, int act,
                 int batches, long sA, long sB, long sC, long sRes)
{
  int Malloc = pad32(M);      // A rows guaranteed allocated
  int Nalloc = pad64(N);      // Bt rows guaranteed allocated
  dim3 g((Nalloc + 127) / 128, (Malloc + 127) / 128, batches);
  k_gemm_nt<<<g, 256, 0, st>>>(A, B, C, M, N, K, Malloc, Nalloc, lda, ldb, ldc,
                               bias, biasMode, res, ldres, act, sA, sB, sC, sRes);
}

extern "C" void kernel_launch(void* const* d_in, const int* in_sizes, int n_in,
                              void* d_out, int out_size, void* d_ws, size_t ws_size,
                              hipStream_t stream)
{
  (void)in_sizes; (void)n_in; (void)out_size; (void)ws_size;
  static const int LC[6]  = {512, 1024, 512, 256, 256, 256};
  static const int LH[6]  = {38, 19, 10, 5, 3, 1};
  static const int LMB[6] = {4, 6, 6, 6, 4, 4};
  static const int BOX_OFF[6] = {0, 5776, 7942, 8542, 8692, 8728};
  const int Bn = 4, S = 5, TOTB = 8732;

  const float* Q[6]; const float* SS[6];
  for (int i = 0; i < 6; ++i) { Q[i] = (const float*)d_in[2 * i]; SS[i] = (const float*)d_in[2 * i + 1]; }

  // Params: jax pytree flatten order (dict keys sorted alphabetically)
  int pi = 12;
  auto nxt = [&]() -> const float* { return (const float*)d_in[pi++]; };
  const float* l2w = nxt();
  LevelParams LP[6];
  for (int l = 0; l < 6; ++l) {
    LevelParams& P = LP[l];
    P.conf_b = nxt(); P.conf_w = nxt(); P.fuse_b = nxt(); P.fuse_w = nxt();
    P.last_gb = nxt(); P.last_gw = nxt();
    for (int m = 0; m < 5; ++m) { P.last_wc_w[m] = nxt(); P.last_wc_b[m] = nxt(); }
    for (int L = 0; L < 5; ++L) {
      P.lay_gb[L] = nxt(); P.lay_gw[L] = nxt();
      for (int m = 0; m < 5; ++m) { P.lay_wc_w[L][m] = nxt(); P.lay_wc_b[L][m] = nxt(); }
    }
    P.loc_b = nxt(); P.loc_w = nxt();
    P.g_b = nxt(); P.g_w = nxt(); P.phi_b = nxt(); P.phi_w = nxt();
    P.theta_b = nxt(); P.theta_w = nxt(); P.wz_b = nxt(); P.wz_w = nxt();
  }

  char* wsb = (char*)d_ws;
  size_t cur = 0;
  auto alloc = [&](size_t bytes) -> void* {
    void* p = wsb + cur; cur += (bytes + 255) & ~(size_t)255; return p;
  };

  // L2Norm on q0 (persistent within this launch)
  float* q0n = (float*)alloc((size_t)Bn * 512 * 1444 * 4);
  {
    int tot = Bn * 1444;
    k_l2norm<<<(tot + 255) / 256, 256, 0, stream>>>(Q[0], l2w, q0n, 512, 1444, tot);
  }
  const size_t persist = cur;

  for (int l = 0; l < 6; ++l) {
    cur = persist;                    // scratch reused level-to-level
    const LevelParams& P = LP[l];
    const int c = LC[l], ci = c / 2, H = LH[l], Np = H * H;
    const int NP = pad64(Np);         // position dim padded to 64 (full tiles)
    const int mbx = LMB[l];
    const float* qsrc = (l == 0) ? q0n : Q[l];
    const int BS = Bn * S;

    // ---- bf16 weight staging (K contiguous) ----
    bf16_t* thW = (bf16_t*)alloc((size_t)ci * c * 2);
    bf16_t* phW = (bf16_t*)alloc((size_t)ci * c * 2);
    bf16_t* gW  = (bf16_t*)alloc((size_t)ci * c * 2);
    bf16_t* wzW = (bf16_t*)alloc((size_t)c * ci * 2);
    bf16_t* fuW = (bf16_t*)alloc((size_t)c * c * 2);
    {
      long tw = (long)ci * c;
      k_cast_pad<<<dim3((tw + 255) / 256, 1), 256, 0, stream>>>(P.theta_w, thW, ci, c, c, ci, c, 0, 0);
      k_cast_pad<<<dim3((tw + 255) / 256, 1), 256, 0, stream>>>(P.phi_w,   phW, ci, c, c, ci, c, 0, 0);
      k_cast_pad<<<dim3((tw + 255) / 256, 1), 256, 0, stream>>>(P.g_w,     gW,  ci, c, c, ci, c, 0, 0);
      long tz = (long)c * ci;
      k_cast_pad<<<dim3((tz + 255) / 256, 1), 256, 0, stream>>>(P.wz_w, wzW, c, ci, ci, c, ci, 0, 0);
      long tf = (long)c * c;
      k_cast_pad<<<dim3((tf + 255) / 256, 1), 256, 0, stream>>>(P.fuse_w, fuW, c, c, c, c, c, 0, 0);
    }

    // ---- activation staging: q^T, sup^T as bf16 [NP x c] ----
    bf16_t* qT   = (bf16_t*)alloc((size_t)Bn * NP * c * 2);
    bf16_t* supT = (bf16_t*)alloc((size_t)BS * NP * c * 2);
    k_cast_T_pad<<<dim3(c / 16, NP / 16, Bn), 256, 0, stream>>>(
        qsrc, qT, c, Np, Np, NP, c, (long)c * Np, (long)NP * c);
    k_cast_T_pad<<<dim3(c / 16, NP / 16, BS), 256, 0, stream>>>(
        SS[l], supT, c, Np, Np, NP, c, (long)c * Np, (long)NP * c);

    // ---- theta^T = qT x thW^T  -> (Np x ci) ----
    float*  thetaT  = (float*)alloc((size_t)Bn * NP * ci * 4);
    bf16_t* thetaTb = (bf16_t*)alloc((size_t)Bn * NP * ci * 2);
    gemm(stream, qT, thW, thetaT, Np, ci, c, c, c, ci,
         P.theta_b, 2, 0, 0, 0, Bn, (long)NP * c, 0, (long)NP * ci, 0);
    k_cast_pad<<<dim3(((long)NP * ci + 255) / 256, Bn), 256, 0, stream>>>(
        thetaT, thetaTb, Np, ci, ci, NP, ci, (long)NP * ci, (long)NP * ci);

    // ---- phi^T = supT x phW^T -> (Np x ci) per (b,s) ----
    float*  phiT  = (float*)alloc((size_t)BS * NP * ci * 4);
    bf16_t* phiTb = (bf16_t*)alloc((size_t)BS * NP * ci * 2);
    gemm(stream, supT, phW, phiT, Np, ci, c, c, c, ci,
         P.phi_b, 2, 0, 0, 0, BS, (long)NP * c, 0, (long)NP * ci, 0);
    k_cast_pad<<<dim3(((long)NP * ci + 255) / 256, BS), 256, 0, stream>>>(
        phiT, phiTb, Np, ci, ci, NP, ci, (long)NP * ci, (long)NP * ci);

    // ---- g = gW x sup -> (ci x Np) per (b,s), keys contiguous ----
    float*  gbf  = (float*)alloc((size_t)BS * ci * NP * 4);
    bf16_t* gbfb = (bf16_t*)alloc((size_t)BS * ci * NP * 2);
    gemm(stream, gW, supT, gbf, ci, Np, c, c, c, NP,
         P.g_b, 1, 0, 0, 0, BS, 0, (long)NP * c, (long)ci * NP, 0);
    k_cast_pad<<<dim3(((long)ci * NP + 255) / 256, BS), 256, 0, stream>>>(
        gbf, gbfb, ci, Np, NP, ci, NP, (long)ci * NP, (long)ci * NP);

    // ---- attention scores f = theta^T x phi (Np x Np), batched over (b,s) ----
    float*  fbuf = (float*)alloc((size_t)BS * NP * NP * 4);
    bf16_t* Pbuf = (bf16_t*)alloc((size_t)BS * NP * NP * 2);
    for (int b = 0; b < Bn; ++b) {
      gemm(stream, thetaTb + (size_t)b * NP * ci, phiTb + (size_t)b * S * NP * ci,
           fbuf + (size_t)b * S * NP * NP, Np, Np, ci, ci, ci, NP,
           0, 0, 0, 0, 0, S, 0, (long)NP * ci, (long)NP * NP, 0);
    }
    k_softmax_bf16<<<dim3(NP, BS), 256, 0, stream>>>(
        fbuf, Pbuf, Np, NP, (long)NP * NP, (long)NP * NP);

    // ---- y = P x g^T -> (Np x ci) ----
    float*  ybuf  = (float*)alloc((size_t)BS * NP * ci * 4);
    bf16_t* ybufb = (bf16_t*)alloc((size_t)BS * NP * ci * 2);
    gemm(stream, Pbuf, gbfb, ybuf, Np, ci, NP, NP, NP, ci,
         0, 0, 0, 0, 0, BS, (long)NP * NP, (long)ci * NP, (long)NP * ci, 0);
    k_cast_pad<<<dim3(((long)NP * ci + 255) / 256, BS), 256, 0, stream>>>(
        ybuf, ybufb, Np, ci, ci, NP, ci, (long)NP * ci, (long)NP * ci);

    // ---- z = wz x y^T + wz_b + q (residual) -> (c x Np) per (b,s) ----
    float* zbuf = (float*)alloc((size_t)BS * c * NP * 4);
    for (int b = 0; b < Bn; ++b) {
      gemm(stream, wzW, ybufb + (size_t)b * S * NP * ci, zbuf + (size_t)b * S * c * NP,
           c, Np, ci, ci, ci, NP,
           P.wz_b, 1, qsrc + (size_t)b * c * Np, Np, 0,
           S, 0, (long)NP * ci, (long)c * NP, 0);
    }

    // ---- GNN gates from support GAP features ----
    float* gap   = (float*)alloc((size_t)BS * c * 4);
    float* gates = (float*)alloc((size_t)Bn * S * 4);
    float* xbuf  = (float*)alloc((size_t)Bn * 5 * 1280 * 4);
    float* xabuf = (float*)alloc((size_t)Bn * 5 * 1280 * 4);
    k_gap<<<(BS * c + 255) / 256, 256, 0, stream>>>(SS[l], gap, c, Np, BS * c);
    GnnP GP;
    for (int L = 0; L < 5; ++L) {
      GP.gw[L] = P.lay_gw[L]; GP.gb[L] = P.lay_gb[L];
      for (int m = 0; m < 5; ++m) { GP.wc_w[L][m] = P.lay_wc_w[L][m]; GP.wc_b[L][m] = P.lay_wc_b[L][m]; }
    }
    GP.gw[5] = P.last_gw; GP.gb[5] = P.last_gb;
    for (int m = 0; m < 5; ++m) { GP.wc_w[5][m] = P.last_wc_w[m]; GP.wc_b[5][m] = P.last_wc_b[m]; }
    k_gnn<<<Bn, 256, 0, stream>>>(gap, gates, xbuf, xabuf, c, 1280, GP);

    // ---- gated sum over shots ----
    float* fused = (float*)alloc((size_t)Bn * c * NP * 4);
    k_fused_sum<<<dim3(((long)c * Np + 255) / 256, Bn), 256, 0, stream>>>(
        zbuf, gates, fused, c, Np, NP);

    // ---- fuse conv1x1 + relu ----
    bf16_t* fusedT = (bf16_t*)alloc((size_t)Bn * NP * c * 2);
    float*  ffo    = (float*)alloc((size_t)Bn * c * NP * 4);
    k_cast_T_pad<<<dim3(c / 16, NP / 16, Bn), 256, 0, stream>>>(
        fused, fusedT, c, Np, NP, NP, c, (long)c * NP, (long)NP * c);
    gemm(stream, fuW, fusedT, ffo, c, Np, c, c, c, NP,
         P.fuse_b, 1, 0, 0, 1, Bn, 0, (long)NP * c, (long)c * NP, 0);

    // ---- SSD heads (conv3x3) straight into d_out ----
    float* out_loc  = (float*)d_out;
    float* out_conf = (float*)d_out + (size_t)Bn * TOTB * 4;
    {
      int Ol = mbx * 4, tot = Ol * Np;
      int gx = (tot + 255) / 256; if (gx > 4096) gx = 4096;
      k_head<<<dim3(gx, Bn), 256, 0, stream>>>(ffo, (long)c * NP, P.loc_w, P.loc_b,
                                               out_loc, c, NP, H, H, mbx, 4,
                                               BOX_OFF[l], TOTB);
      int Oc = mbx * 21; tot = Oc * Np;
      gx = (tot + 255) / 256; if (gx > 4096) gx = 4096;
      k_head<<<dim3(gx, Bn), 256, 0, stream>>>(ffo, (long)c * NP, P.conf_w, P.conf_b,
                                               out_conf, c, NP, H, H, mbx, 21,
                                               BOX_OFF[l], TOTB);
    }
  }
}
